// GroupedQueryAttention_9457517985810
// MI455X (gfx1250) — matrix operand order repaired and checked by reading, not compiled, not run on hardware
//
#include <hip/hip_runtime.h>
#include <hip/hip_bf16.h>

typedef __attribute__((ext_vector_type(16))) __bf16 v16bf;
typedef __attribute__((ext_vector_type(8)))  __bf16 v8bf;
typedef __attribute__((ext_vector_type(8)))  float  v8f;

constexpr int Bn   = 4;
constexpr int Ssz  = 2048;
constexpr int DM   = 2048;   // d_model
constexpr int NH   = 16;     // query heads
constexpr int NKV  = 4;      // kv heads
constexpr int DK   = 128;    // head dim
constexpr int ROWS = Bn * Ssz;     // 8192
constexpr int DKV  = NKV * DK;     // 512

#define DEVI __device__ __forceinline__

DEVI v8f wmma_bf16(v16bf a, v16bf b, v8f c) {
  return __builtin_amdgcn_wmma_f32_16x16x32_bf16(
      /*neg_a=*/false, a, /*neg_b=*/false, b,
      /*c_mod=*/(short)0, c, /*reuse_a=*/false, /*reuse_b=*/false);
}

// A-matrix 16x32 bf16 fragment: caller passes pointer to row m (=lane&15) at k0.
// lanes 0-15: K {0..7, 16..23}; lanes 16-31: K {8..15, 24..31}
DEVI v16bf load_a_frag(const __bf16* rowk0, int lane) {
  const int kh = (lane >> 4) << 3;           // 0 or 8
  v8bf lo = *(const v8bf*)(rowk0 + kh);
  v8bf hi = *(const v8bf*)(rowk0 + 16 + kh);
  v16bf r;
#pragma unroll
  for (int i = 0; i < 8; ++i) { r[i] = lo[i]; r[i + 8] = hi[i]; }
  return r;
}

// B-matrix 32x16 bf16 fragment: caller passes pointer to B^T row n (=lane&15) at k0.
// lanes 0-15: K {0..7, 16..23}; lanes 16-31: K {8..15, 24..31}
DEVI v16bf load_b_frag(const __bf16* rowk0, int lane) {
  const int kh = (lane >> 4) << 3;           // 0 or 8
  v8bf lo = *(const v8bf*)(rowk0 + kh);
  v8bf hi = *(const v8bf*)(rowk0 + 16 + kh);
  v16bf r;
#pragma unroll
  for (int i = 0; i < 8; ++i) { r[i] = lo[i]; r[i + 8] = hi[i]; }
  return r;
}

// ---------------- conversion kernels ----------------

__global__ void k_cvt(const float* __restrict__ in, __bf16* __restrict__ out, int n) {
  int i = blockIdx.x * blockDim.x + threadIdx.x;
  if (i < n) out[i] = (__bf16)in[i];
}

// W [Kd x N] fp32  ->  Wt [N x Kd] bf16   (tiled 32x32 through LDS)
__global__ void k_cvt_tr(const float* __restrict__ W, __bf16* __restrict__ Wt,
                         int Kd, int N) {
  __shared__ float tile[32][33];
  const int tx = threadIdx.x, ty = threadIdx.y;     // block (32, 8)
  const int k0 = blockIdx.y * 32, n0 = blockIdx.x * 32;
#pragma unroll
  for (int i = 0; i < 32; i += 8)
    tile[ty + i][tx] = W[(size_t)(k0 + ty + i) * N + n0 + tx];
  __syncthreads();
#pragma unroll
  for (int i = 0; i < 32; i += 8)
    Wt[(size_t)(n0 + ty + i) * Kd + k0 + tx] = (__bf16)tile[tx][ty + i];
}

// ---------------- GEMM: C[M,N] = A[M,Kd] * Bt[N,Kd]^T ----------------
// block = 256 threads = 8 waves in 2x4; block tile 64x256; wave tile 32x64.
template <bool F32OUT>
__global__ __launch_bounds__(256, 1)
void k_gemm(const __bf16* __restrict__ A, const __bf16* __restrict__ Bt,
            void* __restrict__ C, int M, int N, int Kd) {
  const int lane = threadIdx.x & 31;
  const int wid  = threadIdx.x >> 5;
  const int wm = wid >> 2, wn = wid & 3;
  const int row0 = blockIdx.y * 64 + wm * 32;
  const int col0 = blockIdx.x * 256 + wn * 64;
  const int am = lane & 15;

  const __bf16* arow[2];
  const __bf16* brow[4];
  arow[0] = A + (size_t)(row0 + am) * Kd;
  arow[1] = arow[0] + (size_t)16 * Kd;
#pragma unroll
  for (int j = 0; j < 4; ++j) brow[j] = Bt + (size_t)(col0 + j * 16 + am) * Kd;

  v8f acc[2][4] = {};
  for (int k0 = 0; k0 < Kd; k0 += 32) {
    __builtin_prefetch(arow[0] + k0 + 256, 0, 1);
    __builtin_prefetch(brow[0] + k0 + 256, 0, 1);
    v16bf a[2], b[4];
#pragma unroll
    for (int i = 0; i < 2; ++i) a[i] = load_a_frag(arow[i] + k0, lane);
#pragma unroll
    for (int j = 0; j < 4; ++j) b[j] = load_b_frag(brow[j] + k0, lane);
#pragma unroll
    for (int i = 0; i < 2; ++i)
#pragma unroll
      for (int j = 0; j < 4; ++j) acc[i][j] = wmma_bf16(a[i], b[j], acc[i][j]);
  }

  const int n = am, mb = (lane >> 4) << 3;
#pragma unroll
  for (int i = 0; i < 2; ++i)
#pragma unroll
    for (int j = 0; j < 4; ++j) {
      const int r0 = row0 + i * 16 + mb;
      const int cc = col0 + j * 16 + n;
#pragma unroll
      for (int v = 0; v < 8; ++v) {
        if constexpr (F32OUT)
          ((float*)C)[(size_t)(r0 + v) * N + cc] = acc[i][j][v];
        else
          ((__bf16*)C)[(size_t)(r0 + v) * N + cc] = (__bf16)acc[i][j][v];
      }
    }
}

// ---------------- RoPE (interleaved pairs, base 10000), in place on bf16 ----------------
__global__ void k_rope(__bf16* __restrict__ p, int nh) {
  const int half = DK / 2;
  const int total = ROWS * nh * half;
  int idx = blockIdx.x * blockDim.x + threadIdx.x;
  if (idx >= total) return;
  const int i   = idx % half;
  const int t   = idx / half;
  const int h   = t % nh;
  const int row = t / nh;
  const int s   = row % Ssz;
  // inv_freq = 10000^(-2i/128)
  const float ang = (float)s * __expf(-(float)(2 * i) * (9.210340371976184f / (float)DK));
  float cs, sn;
  __sincosf(ang, &sn, &cs);
  __bf16* q = p + (size_t)row * (nh * DK) + h * DK + 2 * i;
  const float x0 = (float)q[0], x1 = (float)q[1];
  q[0] = (__bf16)(x0 * cs - x1 * sn);
  q[1] = (__bf16)(x1 * cs + x0 * sn);
}

// ---------------- V transpose: V[b*S+s, kv*128+d] -> Vt[((b*4+kv)*128+d)*S + s] ----------------
__global__ void k_vtr(const __bf16* __restrict__ V, __bf16* __restrict__ Vt) {
  int idx = blockIdx.x * blockDim.x + threadIdx.x;
  if (idx >= ROWS * DKV) return;
  const int c = idx % DKV, r = idx / DKV;
  const int b = r / Ssz, s = r % Ssz;
  const int kv = c / DK, d = c % DK;
  Vt[((size_t)(b * NKV + kv) * DK + d) * Ssz + s] = V[idx];
}

// ---------------- causal flash attention, one wave per 16-row Q tile ----------------
__global__ __launch_bounds__(256, 1)
void k_attn(const __bf16* __restrict__ Q, const __bf16* __restrict__ K,
            const __bf16* __restrict__ Vt, __bf16* __restrict__ O) {
  __shared__ __align__(16) __bf16 plds[8][16 * 32];   // per-wave P scratch
  const int lane = threadIdx.x & 31;
  const int wid  = threadIdx.x >> 5;
  const int gw   = blockIdx.x * 8 + wid;
  const int qt = gw & 127;               // S/16 = 128 tiles
  const int h  = (gw >> 7) & 15;
  const int b  = gw >> 11;
  const int kv = h >> 2;                 // GQA: head h uses kv head h/4
  const int q0 = qt * 16;
  const int am = lane & 15;
  const int mb = (lane >> 4) << 3;
  const float scale = 0.08838834764831845f;  // 1/sqrt(128)

  // Q fragments for this tile: 16x128 = 4 A-frags of 16x32
  const __bf16* qrow = Q + ((size_t)(b * Ssz + q0 + am) * DM + h * DK);
  v16bf qa[4];
#pragma unroll
  for (int d = 0; d < 4; ++d) qa[d] = load_a_frag(qrow + d * 32, lane);

  v8f o[8];
#pragma unroll
  for (int t = 0; t < 8; ++t) o[t] = v8f{};
  float mrun[8], lrun[8];
#pragma unroll
  for (int v = 0; v < 8; ++v) { mrun[v] = -3.0e38f; lrun[v] = 0.f; }

  const int nblk = (q0 + 47) >> 5;       // key blocks of 32, through the diagonal
  for (int kb = 0; kb < nblk; ++kb) {
    const int s0 = kb * 32;
    // ---- scores: two 16x16 tiles, contract over d=128 in 4 WMMA steps each
    v8f sc[2];
#pragma unroll
    for (int j = 0; j < 2; ++j) {
      v8f a{};
      const __bf16* krow = K + ((size_t)(b * Ssz + s0 + j * 16 + am) * DKV + kv * DK);
#pragma unroll
      for (int d = 0; d < 4; ++d) {
        v16bf bf = load_b_frag(krow + d * 32, lane);
        a = wmma_bf16(qa[d], bf, a);
      }
      sc[j] = a;
    }
    // ---- scale + causal mask
#pragma unroll
    for (int j = 0; j < 2; ++j)
#pragma unroll
      for (int v = 0; v < 8; ++v) {
        const float val = sc[j][v] * scale;
        sc[j][v] = ((s0 + j * 16 + am) <= (q0 + mb + v)) ? val : -3.0e38f;
      }
    // ---- online softmax stats (rows live across 16-lane halves)
    float alpha[8];
#pragma unroll
    for (int v = 0; v < 8; ++v) {
      float r = fmaxf(sc[0][v], sc[1][v]);
#pragma unroll
      for (int off = 1; off < 16; off <<= 1) r = fmaxf(r, __shfl_xor(r, off, 32));
      const float mn = fmaxf(mrun[v], r);
      alpha[v] = __expf(mrun[v] - mn);
      mrun[v] = mn;
    }
#pragma unroll
    for (int v = 0; v < 8; ++v) {
      const float p0 = __expf(sc[0][v] - mrun[v]);
      const float p1 = __expf(sc[1][v] - mrun[v]);
      sc[0][v] = p0; sc[1][v] = p1;
      float r = p0 + p1;
#pragma unroll
      for (int off = 1; off < 16; off <<= 1) r += __shfl_xor(r, off, 32);
      lrun[v] = lrun[v] * alpha[v] + r;
    }
    // ---- P (16x32) -> bf16 in LDS (C-layout write), reread in A-layout
#pragma unroll
    for (int j = 0; j < 2; ++j)
#pragma unroll
      for (int v = 0; v < 8; ++v)
        plds[wid][(mb + v) * 32 + j * 16 + am] = (__bf16)sc[j][v];
    asm volatile("s_wait_dscnt 0" ::: "memory");
    v16bf pf;
    {
      const int kh = (lane >> 4) << 3;
      v8bf lo = *(const v8bf*)(&plds[wid][am * 32 + kh]);
      v8bf hi = *(const v8bf*)(&plds[wid][am * 32 + 16 + kh]);
#pragma unroll
      for (int i = 0; i < 8; ++i) { pf[i] = lo[i]; pf[i + 8] = hi[i]; }
    }
    // ---- rescale running output, accumulate P @ V (8 d-tiles of 16)
#pragma unroll
    for (int t = 0; t < 8; ++t) {
#pragma unroll
      for (int v = 0; v < 8; ++v) o[t][v] *= alpha[v];
      const __bf16* vrow = Vt + (((size_t)(b * NKV + kv) * DK + t * 16 + am) * Ssz + s0);
      v16bf bf = load_b_frag(vrow, lane);
      o[t] = wmma_bf16(pf, bf, o[t]);
    }
  }
  // ---- normalize and write out (bf16, heads concatenated)
#pragma unroll
  for (int t = 0; t < 8; ++t)
#pragma unroll
    for (int v = 0; v < 8; ++v)
      O[(size_t)(b * Ssz + q0 + mb + v) * DM + h * DK + t * 16 + am] =
          (__bf16)(o[t][v] * (1.0f / lrun[v]));
}

// ---------------- host launcher ----------------
extern "C" void kernel_launch(void* const* d_in, const int* in_sizes, int n_in,
                              void* d_out, int out_size, void* d_ws, size_t ws_size,
                              hipStream_t stream) {
  const float* x  = (const float*)d_in[0];
  const float* Wq = (const float*)d_in[1];
  const float* Wk = (const float*)d_in[2];
  const float* Wv = (const float*)d_in[3];
  const float* Wo = (const float*)d_in[4];

  char* p = (char*)d_ws;
  auto alloc = [&](size_t bytes) {
    char* r = p; p += (bytes + 255) & ~size_t(255); return r;
  };
  __bf16* xb  = (__bf16*)alloc((size_t)ROWS * DM * 2);
  __bf16* Qb  = (__bf16*)alloc((size_t)ROWS * DM * 2);
  __bf16* Kb  = (__bf16*)alloc((size_t)ROWS * DKV * 2);
  __bf16* Vb  = (__bf16*)alloc((size_t)ROWS * DKV * 2);
  __bf16* Vtb = (__bf16*)alloc((size_t)ROWS * DKV * 2);
  __bf16* Ab  = (__bf16*)alloc((size_t)ROWS * DM * 2);
  __bf16* Wqt = (__bf16*)alloc((size_t)DM * DM * 2);
  __bf16* Wkt = (__bf16*)alloc((size_t)DKV * DM * 2);
  __bf16* Wvt = (__bf16*)alloc((size_t)DKV * DM * 2);
  __bf16* Wot = (__bf16*)alloc((size_t)DM * DM * 2);

  const int n = ROWS * DM;
  k_cvt<<<(n + 255) / 256, 256, 0, stream>>>(x, xb, n);

  dim3 tb(32, 8);
  k_cvt_tr<<<dim3(DM / 32, DM / 32), tb, 0, stream>>>(Wq, Wqt, DM, DM);
  k_cvt_tr<<<dim3(DKV / 32, DM / 32), tb, 0, stream>>>(Wk, Wkt, DM, DKV);
  k_cvt_tr<<<dim3(DKV / 32, DM / 32), tb, 0, stream>>>(Wv, Wvt, DM, DKV);
  k_cvt_tr<<<dim3(DM / 32, DM / 32), tb, 0, stream>>>(Wo, Wot, DM, DM);

  k_gemm<false><<<dim3(DM / 256, ROWS / 64), 256, 0, stream>>>(xb, Wqt, Qb, ROWS, DM, DM);
  k_gemm<false><<<dim3(DKV / 256, ROWS / 64), 256, 0, stream>>>(xb, Wkt, Kb, ROWS, DKV, DM);
  k_gemm<false><<<dim3(DKV / 256, ROWS / 64), 256, 0, stream>>>(xb, Wvt, Vb, ROWS, DKV, DM);

  const int nq = ROWS * NH * (DK / 2);
  k_rope<<<(nq + 255) / 256, 256, 0, stream>>>(Qb, NH);
  const int nk = ROWS * NKV * (DK / 2);
  k_rope<<<(nk + 255) / 256, 256, 0, stream>>>(Kb, NKV);

  const int nv = ROWS * DKV;
  k_vtr<<<(nv + 255) / 256, 256, 0, stream>>>(Vb, Vtb);

  k_attn<<<(Bn * NH * (Ssz / 16)) / 8, 256, 0, stream>>>(Qb, Kb, Vtb, Ab);

  k_gemm<true><<<dim3(DM / 256, ROWS / 64), 256, 0, stream>>>(Ab, Wot, (float*)d_out,
                                                              ROWS, DM, DM);
}
